// AttentionModel_39290360824605
// MI455X (gfx1250) — matrix operand order repaired and checked by reading, not compiled, not run on hardware
//
#include <hip/hip_runtime.h>
#include <hip/hip_bf16.h>
#include <stdint.h>

// ---------------------------------------------------------------------------
// CDNA5 / gfx1250: wave32, WMMA bf16 16x16x32 (f32 accumulate),
// async global->LDS tile staging (ASYNCcnt) when the toolchain exposes it.
// ---------------------------------------------------------------------------
typedef __attribute__((ext_vector_type(16))) __bf16 v16bf;
typedef __attribute__((ext_vector_type(8)))  float  v8f;
typedef __attribute__((ext_vector_type(4)))  int    v4i;

#define DEVINL __device__ __forceinline__

struct alignas(16) U4 { uint32_t x, y, z, w; };
union Frag { v16bf v; U4 u[2]; };

#if __has_builtin(__builtin_amdgcn_global_load_async_to_lds_b128)
#define ASYNC_COPY 1
#else
#define ASYNC_COPY 0
#endif

#if ASYNC_COPY
template <int OFF>
DEVINL void async_b128(const uint16_t* g, uint16_t* l) {
  // Builtin takes generic int4* pointers (probe-verified via diagnostics).
  // IOFFSET is added to BOTH the global and the LDS address (ISA 10.7), so
  // one immediate advances both sides of the copy.
  __builtin_amdgcn_global_load_async_to_lds_b128((v4i*)g, (v4i*)l, OFF, 0);
}
DEVINL void wait_async4() {
#if __has_builtin(__builtin_amdgcn_s_wait_asynccnt)
  __builtin_amdgcn_s_wait_asynccnt(4);
#else
  asm volatile("s_wait_asynccnt 0x4" ::: "memory");
#endif
}
DEVINL void wait_async0() {
#if __has_builtin(__builtin_amdgcn_s_wait_asynccnt)
  __builtin_amdgcn_s_wait_asynccnt(0);
#else
  asm volatile("s_wait_asynccnt 0x0" ::: "memory");
#endif
}
#endif

DEVINL uint16_t f2bf(float f) {                 // round-to-nearest-even fp32->bf16
  union { float f; uint32_t u; } x; x.f = f;
  uint32_t r = x.u + 0x7FFFu + ((x.u >> 16) & 1u);
  return (uint16_t)(r >> 16);
}
DEVINL float bf2f(uint16_t h) {
  union { uint32_t u; float f; } x; x.u = ((uint32_t)h) << 16;
  return x.f;
}
DEVINL float sigf(float x) { return 1.0f / (1.0f + __expf(-x)); }

// A-fragment: 16x32 bf16, row m = m0 + lane%16; lane group g=lane/16 holds
// K run [k0+16g, +16)  (two 16B contiguous loads)
DEVINL v16bf load_a_global(const uint16_t* A, long lda, int m0, int k0, unsigned lane) {
  int m = m0 + (int)(lane & 15u);
  int g = (int)(lane >> 4u);
  const uint16_t* p = A + (long)m * lda + k0 + 16 * g;
  Frag r;
  r.u[0] = *(const U4*)p;
  r.u[1] = *(const U4*)(p + 8);
  return r.v;
}
// B-fragment: 32x16 bf16 where B[k][n] = W[n][k]; lane n holds 16 contiguous
// K values [k0+16g, +16) of row n of W (two 16B contiguous loads)
DEVINL Frag load_b_global(const uint16_t* W, long ldb, int n0, int k0, unsigned lane) {
  int n = n0 + (int)(lane & 15u);
  int g = (int)(lane >> 4u);
  const uint16_t* p = W + (long)n * ldb + k0 + 16 * g;
  Frag r;
  r.u[0] = *(const U4*)p;
  r.u[1] = *(const U4*)(p + 8);
  return r;
}

DEVINL v8f wmma_bf16(v16bf a, v16bf b, v8f c) {
  return __builtin_amdgcn_wmma_f32_16x16x32_bf16(false, a, false, b, (short)0, c, false, false);
}

// ---------------------------------------------------------------------------
// Elementwise helpers
// ---------------------------------------------------------------------------
__global__ void f2bf_kernel(const float* __restrict__ in, uint16_t* __restrict__ out, long n) {
  long i = (long)blockIdx.x * blockDim.x + threadIdx.x;
  long stride = (long)gridDim.x * blockDim.x;
  for (; i < n; i += stride) out[i] = f2bf(in[i]);
}

__global__ void vecadd_kernel(const float* __restrict__ a, const float* __restrict__ b,
                              float* __restrict__ o, int n) {
  int i = blockIdx.x * blockDim.x + threadIdx.x;
  if (i < n) o[i] = a[i] + b[i];
}

// enc = concat(emb_in[src], pos_emb[pos]) -> bf16 row-major [B,S,2E] and
// transposed [B, 2E, SP=128] (s-padded) for the context GEMM B-fragments.
__global__ void enc_build_kernel(const int* __restrict__ src, const int* __restrict__ pos,
                                 const float* __restrict__ emb_in, const float* __restrict__ pos_emb,
                                 uint16_t* __restrict__ encb, uint16_t* __restrict__ encT) {
  int bs = blockIdx.x;            // B*S = 3200
  int b = bs / 100, s = bs % 100;
  int wrd = src[b * 100 + s];
  int pp  = pos[b * 100 + s];
  for (int c = threadIdx.x; c < 1024; c += blockDim.x) {
    float v = (c < 512) ? emb_in[(long)wrd * 512 + c] : pos_emb[(long)pp * 512 + (c - 512)];
    uint16_t h = f2bf(v);
    encb[((long)b * 100 + s) * 1024 + c] = h;
    encT[((long)b * 1024 + c) * 128 + s] = h;
  }
}

// mean over S (for h0 GEMM input) + zero the encT padding columns s in [100,128)
__global__ void mean_kernel(const uint16_t* __restrict__ encb, uint16_t* __restrict__ meanb,
                            uint16_t* __restrict__ encT) {
  int b = blockIdx.x;
  for (int c = threadIdx.x; c < 1024; c += blockDim.x) {
    float sum = 0.f;
    for (int s = 0; s < 100; ++s) sum += bf2f(encb[((long)b * 100 + s) * 1024 + c]);
    meanb[b * 1024 + c] = f2bf(sum * 0.01f);
    for (int s = 100; s < 128; ++s) encT[((long)b * 1024 + c) * 128 + s] = 0;
  }
}

// h0 -> c0 (fp32), hA[0] (bf16), queries[:,0,:] (bf16)
__global__ void hidden_init_kernel(const float* __restrict__ hidden, float* __restrict__ cbuf,
                                   uint16_t* __restrict__ hA0, uint16_t* __restrict__ qbf) {
  int b = blockIdx.x;
  for (int c = threadIdx.x; c < 1024; c += blockDim.x) {
    float h = hidden[b * 1024 + c];
    cbuf[b * 1024 + c] = h;
    uint16_t hb = f2bf(h);
    hA0[b * 1024 + c] = hb;
    qbf[((long)b * 64) * 1024 + c] = hb;
  }
}

// x_bf16[t][b][e] = emb_out[tgt[b][t]][e]   (rows ordered (t,b) for the Xg GEMM)
__global__ void xgather_kernel(const int* __restrict__ tgt, const float* __restrict__ emb_out,
                               uint16_t* __restrict__ xbf) {
  int tb = blockIdx.x;            // T*B = 2048
  int t = tb / 32, b = tb % 32;
  int wrd = tgt[b * 64 + t];
  for (int c = threadIdx.x; c < 512; c += blockDim.x)
    xbf[((long)t * 32 + b) * 512 + c] = f2bf(emb_out[(long)wrd * 512 + c]);
}

// ---------------------------------------------------------------------------
// Generic tiled WMMA GEMM:  C[M,N] (f32) = A[M,K](bf16) * B[N,K](bf16)^T + bias
// 256 threads = 8 waves; 128x128 block tile. Wave (wm=wave/2, wn=wave&1)
// computes 2 m-tiles x 4 n-tiles: B-fragment reused by 2 WMMAs -> only
// 12 ds_load_b128 per 8 WMMAs. B-fragments are software-pipelined so each
// WMMA waits on loads issued one step earlier, not its own.
// Double-buffered LDS tiles; async global->LDS staging when available.
// OOB rows/cols are CLAMPED (not predicated): a clamped A/B row only feeds
// C rows/cols that are never stored, so no exec-mask churn in the hot loop.
// ---------------------------------------------------------------------------
__global__ void gemm_bt_kernel(const uint16_t* __restrict__ A, int lda,
                               const uint16_t* __restrict__ B, int ldb,
                               const float* __restrict__ bias,
                               float* __restrict__ C, int ldc,
                               int M, int N, int K) {
  __shared__ __attribute__((aligned(16))) uint16_t sA[2][128 * 40];
  __shared__ __attribute__((aligned(16))) uint16_t sB[2][128 * 40];

  const int n0b = blockIdx.x * 128;
  const int m0b = blockIdx.y * 128;
  const unsigned lane = threadIdx.x & 31u;
  const int wave = (int)(threadIdx.x >> 5u);
  const int wm = wave >> 1;                     // 0..3 -> m-tiles 2wm, 2wm+1
  const int wn = wave & 1;                      // 0..1 -> n-tiles 4wn..4wn+3

  // staging coordinates: thread -> (row r, 16-half chunk h)
  const int r = (int)(threadIdx.x >> 1u);       // 0..127
  const int h = (int)(threadIdx.x & 1u) * 16;   // 0 or 16 halves
  int mr = m0b + r; if (mr >= M) mr = M - 1;    // clamp (see note above)
  int nr = n0b + r; if (nr >= N) nr = N - 1;
  const uint16_t* gA = A + (long)mr * lda + h;
  const uint16_t* gB = B + (long)nr * ldb + h;

  v8f acc[8];                                   // [i(2)][nt(4)] -> acc[i*4+nt]
#pragma unroll
  for (int i = 0; i < 8; ++i) acc[i] = (v8f)(0.f);

  const int nk = K >> 5;

#if ASYNC_COPY
  // prologue: issue stage 0 into buffer 0
  {
    uint16_t* la = &sA[0][r * 40 + h];
    uint16_t* lb = &sB[0][r * 40 + h];
    async_b128<0>(gA, la);  async_b128<16>(gA, la);
    async_b128<0>(gB, lb);  async_b128<16>(gB, lb);
  }
#endif

  const int g = (int)(lane >> 4u);
  const int l15 = (int)(lane & 15u);

  for (int ks = 0; ks < nk; ++ks) {
    const int cur = ks & 1;

#if ASYNC_COPY
    if (ks + 1 < nk) {
      // prefetch stage ks+1 into the other buffer, then retire current stage
      const uint16_t* pa = gA + (ks + 1) * 32;
      const uint16_t* pb = gB + (ks + 1) * 32;
      uint16_t* la = &sA[cur ^ 1][r * 40 + h];
      uint16_t* lb = &sB[cur ^ 1][r * 40 + h];
      async_b128<0>(pa, la);  async_b128<16>(pa, la);
      async_b128<0>(pb, lb);  async_b128<16>(pb, lb);
      wait_async4();          // oldest 4 (current stage) complete, in order
    } else {
      wait_async0();
    }
    __syncthreads();
#else
    __syncthreads();
    {
      const uint16_t* pa = gA + ks * 32;
      const uint16_t* pb = gB + ks * 32;
      U4 a0 = *(const U4*)pa, a1 = *(const U4*)(pa + 8);
      U4 b0 = *(const U4*)pb, b1 = *(const U4*)(pb + 8);
      *(U4*)&sA[cur][r * 40 + h] = a0; *(U4*)&sA[cur][r * 40 + h + 8] = a1;
      *(U4*)&sB[cur][r * 40 + h] = b0; *(U4*)&sB[cur][r * 40 + h + 8] = b1;
    }
    __syncthreads();
#endif

    // ---- compute from buffer `cur` ----
    const int am0 = (2 * wm + 0) * 16 + l15;
    const int am1 = (2 * wm + 1) * 16 + l15;
    Frag a0, a1;
    a0.u[0] = *(const U4*)&sA[cur][am0 * 40 + 16 * g];
    a0.u[1] = *(const U4*)&sA[cur][am0 * 40 + 16 * g + 8];
    a1.u[0] = *(const U4*)&sA[cur][am1 * 40 + 16 * g];
    a1.u[1] = *(const U4*)&sA[cur][am1 * 40 + 16 * g + 8];

    // software-pipelined B fragments (n-tiles 4wn .. 4wn+3)
    Frag bc;
    {
      const int bn = (4 * wn) * 16 + l15;
      bc.u[0] = *(const U4*)&sB[cur][bn * 40 + 16 * g];
      bc.u[1] = *(const U4*)&sB[cur][bn * 40 + 16 * g + 8];
    }
#pragma unroll
    for (int nt = 0; nt < 4; ++nt) {
      Frag bnx;
      if (nt < 3) {
        const int bn = (4 * wn + nt + 1) * 16 + l15;
        bnx.u[0] = *(const U4*)&sB[cur][bn * 40 + 16 * g];
        bnx.u[1] = *(const U4*)&sB[cur][bn * 40 + 16 * g + 8];
      }
      acc[nt]     = wmma_bf16(a0.v, bc.v, acc[nt]);
      acc[4 + nt] = wmma_bf16(a1.v, bc.v, acc[4 + nt]);
      if (nt < 3) bc = bnx;
    }

#if ASYNC_COPY
    __syncthreads();   // all waves done reading `cur` before it is refilled
#endif
  }

  // C/D layout: lane = n%16 + 16*(m/8), vgpr r = m%8
#pragma unroll
  for (int i = 0; i < 2; ++i) {
#pragma unroll
    for (int nt = 0; nt < 4; ++nt) {
      int n = n0b + (4 * wn + nt) * 16 + l15;
      if (n >= N) continue;
      float bv = bias ? bias[n] : 0.f;
      int mbase = m0b + (2 * wm + i) * 16 + g * 8;
#pragma unroll
      for (int rr = 0; rr < 8; ++rr) {
        int m = mbase + rr;
        if (m < M) C[(long)m * ldc + n] = acc[i * 4 + nt][rr] + bv;
      }
    }
  }
}

// ---------------------------------------------------------------------------
// One LSTM timestep. 16 blocks x 128 threads (4 waves); block j owns h-columns
// [64j,64j+64); wave w computes gate section w (i/f/g/o) with WMMA:
// gates = Xg[t] + h_prev * W_hh^T ; then fused elementwise cell update.
// W_hh B-fragments are software-pipelined across the K loop.
// ---------------------------------------------------------------------------
__global__ void lstm_step_kernel(const uint16_t* __restrict__ hA_in,   // [32][1024] bf16
                                 const uint16_t* __restrict__ Whh,     // [4096][1024] bf16
                                 const float* __restrict__ Xg_t,       // [32][4096] f32
                                 float* __restrict__ cbuf,             // [32][1024] f32
                                 uint16_t* __restrict__ hA_out,        // [32][1024] bf16
                                 uint16_t* __restrict__ qbf,           // [32][64][1024] bf16
                                 uint16_t* __restrict__ comb,          // [2048][2048] bf16
                                 int t) {
  __shared__ float gLDS[4][32][64];   // 32 KB
  const unsigned lane = threadIdx.x & 31u;
  const int w = (int)(threadIdx.x >> 5u);       // gate section 0..3
  const int j = blockIdx.x;                     // column block 0..15
  const int nbase = 1024 * w + 64 * j;

  v8f acc[8];                                   // [mt(2)][nt(4)]
#pragma unroll
  for (int i = 0; i < 8; ++i) acc[i] = (v8f)(0.f);

  Frag bc = load_b_global(Whh, 1024, nbase, 0, lane);   // (nt=0, k0=0)
  for (int k0 = 0; k0 < 1024; k0 += 32) {
    v16bf af0 = load_a_global(hA_in, 1024, 0,  k0, lane);
    v16bf af1 = load_a_global(hA_in, 1024, 16, k0, lane);
#pragma unroll
    for (int nt = 0; nt < 4; ++nt) {
      Frag bnx;
      if (nt < 3)            bnx = load_b_global(Whh, 1024, nbase + 16 * (nt + 1), k0, lane);
      else if (k0 < 1024-32) bnx = load_b_global(Whh, 1024, nbase, k0 + 32, lane);
      acc[nt]     = wmma_bf16(af0, bc.v, acc[nt]);
      acc[4 + nt] = wmma_bf16(af1, bc.v, acc[4 + nt]);
      bc = bnx;
    }
  }

  // gates -> LDS (+ Xg)
#pragma unroll
  for (int mt = 0; mt < 2; ++mt) {
#pragma unroll
    for (int nt = 0; nt < 4; ++nt) {
      int col = 16 * nt + (int)(lane & 15u);
      int ng  = nbase + col;
#pragma unroll
      for (int rr = 0; rr < 8; ++rr) {
        int b = 16 * mt + (int)(lane >> 4u) * 8 + rr;
        gLDS[w][b][col] = acc[mt * 4 + nt][rr] + Xg_t[(long)b * 4096 + ng];
      }
    }
  }
  __syncthreads();

  // fused cell update: 2048 elements / 128 threads
  for (int idx = threadIdx.x; idx < 2048; idx += blockDim.x) {
    int b = idx >> 6, col = idx & 63;
    int n = 64 * j + col;
    float gi = gLDS[0][b][col], gf = gLDS[1][b][col];
    float gg = gLDS[2][b][col], go = gLDS[3][b][col];
    float c = cbuf[b * 1024 + n];
    c = sigf(gf) * c + sigf(gi) * tanhf(gg);
    float hh = sigf(go) * tanhf(c);
    cbuf[b * 1024 + n] = c;
    uint16_t hb = f2bf(hh);
    hA_out[b * 1024 + n] = hb;
    comb[((long)(b * 64 + t)) * 2048 + 1024 + n] = hb;
    if (t + 1 < 64) qbf[((long)b * 64 + (t + 1)) * 1024 + n] = hb;
  }
}

// ---------------------------------------------------------------------------
// Attention: one block per batch (32 blocks x 128 threads, 4 waves).
// energy = Q*enc^T (WMMA, K=1024) -> masked softmax (denominator of the
// unmasked softmax cancels in the renormalization) -> context = align*enc
// (WMMA, K=SP=128), scaled by 1/S, stored bf16 into combined[:, :1024].
// ---------------------------------------------------------------------------
__global__ void attention_kernel(const uint16_t* __restrict__ qbf,   // [32][64][1024]
                                 const uint16_t* __restrict__ encb,  // [32][100][1024]
                                 const uint16_t* __restrict__ encT,  // [32][1024][128]
                                 const int* __restrict__ slen,
                                 uint16_t* __restrict__ comb) {
  __shared__ float    eLDS[64][128];                             // 32 KB
  __shared__ __attribute__((aligned(16))) uint16_t aLDS[64][128];// 16 KB
  const int b = blockIdx.x;
  const unsigned lane = threadIdx.x & 31u;
  const int w = (int)(threadIdx.x >> 5u);    // m-tile (16 query rows)
  const int g = (int)(lane >> 4u);

  // ---- energy (B-fragments software-pipelined, zeros for s >= 100) ----
  v8f acc[8];
#pragma unroll
  for (int i = 0; i < 8; ++i) acc[i] = (v8f)(0.f);

  auto load_enc = [&](int nt, int k0) -> Frag {
    Frag f;
    int s = 16 * nt + (int)(lane & 15u);
    if (s < 100) {
      const uint16_t* p = encb + ((long)b * 100 + s) * 1024 + k0 + 16 * g;
      f.u[0] = *(const U4*)p; f.u[1] = *(const U4*)(p + 8);
    } else {
      U4 z = {0, 0, 0, 0}; f.u[0] = z; f.u[1] = z;
    }
    return f;
  };

  Frag bc = load_enc(0, 0);
  for (int k0 = 0; k0 < 1024; k0 += 32) {
    v16bf af = load_a_global(qbf + (long)b * 64 * 1024, 1024, 16 * w, k0, lane);
#pragma unroll
    for (int nt = 0; nt < 8; ++nt) {
      Frag bnx;
      if (nt < 7)            bnx = load_enc(nt + 1, k0);
      else if (k0 < 1024-32) bnx = load_enc(0, k0 + 32);
      acc[nt] = wmma_bf16(af, bc.v, acc[nt]);
      bc = bnx;
    }
  }
#pragma unroll
  for (int nt = 0; nt < 8; ++nt)
#pragma unroll
    for (int rr = 0; rr < 8; ++rr)
      eLDS[16 * w + g * 8 + rr][16 * nt + (int)(lane & 15u)] = acc[nt][rr];
  __syncthreads();

  // ---- masked softmax per query row ----
  if (threadIdx.x < 64) {
    int tq = threadIdx.x;
    int len = slen[b];                                // in [50,100]
    float mx = -3.0e38f;
    for (int s = 0; s < len; ++s) mx = fmaxf(mx, eLDS[tq][s]);
    float sum = 0.f;
    for (int s = 0; s < len; ++s) { float e = __expf(eLDS[tq][s] - mx); eLDS[tq][s] = e; sum += e; }
    float inv = 1.f / sum;
    for (int s = 0; s < 128; ++s)
      aLDS[tq][s] = (s < len) ? f2bf(eLDS[tq][s] * inv) : (uint16_t)0;
  }
  __syncthreads();

  // ---- context ----
  for (int nt2 = 0; nt2 < 64; ++nt2) {
    v8f c = (v8f)(0.f);
#pragma unroll
    for (int kk = 0; kk < 4; ++kk) {
      int k0 = kk * 32;
      int am = 16 * w + (int)(lane & 15u);
      Frag af;
      af.u[0] = *(const U4*)&aLDS[am][k0 + 16 * g];
      af.u[1] = *(const U4*)&aLDS[am][k0 + 16 * g + 8];
      int hc = 16 * nt2 + (int)(lane & 15u);
      const uint16_t* p = encT + ((long)b * 1024 + hc) * 128 + k0 + 16 * g;
      Frag bfr;
      bfr.u[0] = *(const U4*)p; bfr.u[1] = *(const U4*)(p + 8);
      c = wmma_bf16(af.v, bfr.v, c);
    }
    int n = 16 * nt2 + (int)(lane & 15u);
#pragma unroll
    for (int rr = 0; rr < 8; ++rr) {
      int tq = 16 * w + g * 8 + rr;
      comb[((long)(b * 64 + tq)) * 2048 + n] = f2bf(c[rr] * 0.01f);  // /S, S=100
    }
  }
}

// ---------------------------------------------------------------------------
// Host launch
// ---------------------------------------------------------------------------
extern "C" void kernel_launch(void* const* d_in, const int* in_sizes, int n_in,
                              void* d_out, int out_size, void* d_ws, size_t ws_size,
                              hipStream_t stream) {
  (void)in_sizes; (void)n_in; (void)out_size; (void)ws_size;

  const int*   src     = (const int*)d_in[0];
  const int*   slen    = (const int*)d_in[1];
  const int*   pos     = (const int*)d_in[2];
  const int*   tgt     = (const int*)d_in[3];
  const float* emb_in  = (const float*)d_in[5];
  const float* emb_out = (const float*)d_in[6];
  const float* pos_emb = (const float*)d_in[7];
  const float* W_h0    = (const float*)d_in[8];
  const float* b_h0    = (const float*)d_in[9];
  const float* W_ih    = (const float*)d_in[10];
  const float* W_hh    = (const float*)d_in[11];
  const float* b_ih    = (const float*)d_in[12];
  const float* b_hh    = (const float*)d_in[13];
  const float* W_proj  = (const float*)d_in[14];
  const float* b_proj  = (const float*)d_in[15];
  float* out = (float*)d_out;

  size_t off = 0;
  auto alloc = [&](size_t bytes) -> void* {
    void* p = (char*)d_ws + off;
    off += (bytes + 255) & ~(size_t)255;
    return p;
  };
  uint16_t* Wh0b   = (uint16_t*)alloc(1024UL * 1024 * 2);
  uint16_t* Wihb   = (uint16_t*)alloc(4096UL * 512 * 2);
  uint16_t* Whhb   = (uint16_t*)alloc(4096UL * 1024 * 2);
  uint16_t* Wprojb = (uint16_t*)alloc(32000UL * 2048 * 2);
  float*    bias4h = (float*)   alloc(4096UL * 4);
  uint16_t* encb   = (uint16_t*)alloc(32UL * 100 * 1024 * 2);
  uint16_t* encT   = (uint16_t*)alloc(32UL * 1024 * 128 * 2);
  uint16_t* meanb  = (uint16_t*)alloc(32UL * 1024 * 2);
  float*    hidden = (float*)   alloc(32UL * 1024 * 4);
  float*    cbuf   = (float*)   alloc(32UL * 1024 * 4);
  uint16_t* hA     = (uint16_t*)alloc(2UL * 32 * 1024 * 2);
  uint16_t* qbf    = (uint16_t*)alloc(32UL * 64 * 1024 * 2);
  uint16_t* xbf    = (uint16_t*)alloc(64UL * 32 * 512 * 2);
  float*    Xg     = (float*)   alloc(64UL * 32 * 4096 * 4);
  uint16_t* comb   = (uint16_t*)alloc(2048UL * 2048 * 2);

  // 1) fp32 -> bf16 parameter conversion + fused bias
  f2bf_kernel<<<512, 256, 0, stream>>>(W_h0, Wh0b, 1024L * 1024);
  f2bf_kernel<<<512, 256, 0, stream>>>(W_ih, Wihb, 4096L * 512);
  f2bf_kernel<<<1024, 256, 0, stream>>>(W_hh, Whhb, 4096L * 1024);
  f2bf_kernel<<<4096, 256, 0, stream>>>(W_proj, Wprojb, 32000L * 2048);
  vecadd_kernel<<<16, 256, 0, stream>>>(b_ih, b_hh, bias4h, 4096);

  // 2) encoder: embeddings, transpose, mean-pool
  enc_build_kernel<<<3200, 256, 0, stream>>>(src, pos, emb_in, pos_emb, encb, encT);
  mean_kernel<<<32, 256, 0, stream>>>(encb, meanb, encT);

  // 3) hidden = mean(enc) @ W_h0^T + b_h0   (M=32,N=1024,K=1024)
  gemm_bt_kernel<<<dim3(8, 1), 256, 0, stream>>>(meanb, 1024, Wh0b, 1024, b_h0,
                                                 hidden, 1024, 32, 1024, 1024);
  hidden_init_kernel<<<32, 256, 0, stream>>>(hidden, cbuf, hA, qbf);

  // 4) LSTM input GEMM over all timesteps: Xg = x@W_ih^T + (b_ih+b_hh)
  xgather_kernel<<<2048, 256, 0, stream>>>(tgt, emb_out, xbf);
  gemm_bt_kernel<<<dim3(32, 16), 256, 0, stream>>>(xbf, 512, Wihb, 512, bias4h,
                                                   Xg, 4096, 2048, 4096, 512);

  // 5) sequential recurrence
  for (int t = 0; t < 64; ++t) {
    lstm_step_kernel<<<16, 128, 0, stream>>>(hA + (size_t)(t & 1) * 32 * 1024, Whhb,
                                             Xg + (size_t)t * 32 * 4096, cbuf,
                                             hA + (size_t)((t + 1) & 1) * 32 * 1024,
                                             qbf, comb, t);
  }

  // 6) attention -> combined[:, :1024]
  attention_kernel<<<32, 128, 0, stream>>>(qbf, encb, encT, slen, comb);

  // 7) final projection: out = combined @ W_proj^T + b_proj  (268 GFLOP)
  gemm_bt_kernel<<<dim3(250, 16), 256, 0, stream>>>(comb, 2048, Wprojb, 2048, b_proj,
                                                    out, 32000, 2048, 32000, 2048);
}
